// GCNMasker_79568564126473
// MI455X (gfx1250) — matrix operand order repaired and checked by reading, not compiled, not run on hardware
//
#include <hip/hip_runtime.h>
#include <hip/hip_bf16.h>

// ---------------------------------------------------------------------------
// GCNMasker forward for MI455X (gfx1250).
// Dense node-level GEMMs -> v_wmma_f32_16x16x32_bf16, one wave = 16-row band,
// C/16 accumulators per wave (A-fragment reuse). Weights pre-packed into the
// bf16 B-fragment layout (contiguous b128 loads); A fragments loaded as
// float4 (b128) from 16B-aligned rows; K=39 encoder input zero-padded to 64
// so the GEMM is completely branch/predicate free.
// Edge message pass fused (edge-embed GEMM K=10 in LDS + gather + relu +
// scatter-add with zero-skip). BN = 2-pass column stats.
// ---------------------------------------------------------------------------

typedef __attribute__((ext_vector_type(16))) __bf16 v16bf;
typedef __attribute__((ext_vector_type(8)))  float  v8f;

#define EMB 64
#define GRAPHS 64
#define IN_DIM 39

// ---------------- generic helpers ----------------

__global__ void fill_zero(float* __restrict__ p, int n) {
    int i = blockIdx.x * blockDim.x + threadIdx.x;
    if (i < n) p[i] = 0.0f;
}

__global__ void copy_buf(float* __restrict__ d, const float* __restrict__ s, int n) {
    int i = blockIdx.x * blockDim.x + threadIdx.x;
    if (i < n) d[i] = s[i];
}

// ---------------- pad x[N,39] -> xp[N,64] (zeros beyond 39) ----------------
__global__ void pad_x_kernel(const float* __restrict__ x, float* __restrict__ xp, int N) {
    int idx = blockIdx.x * blockDim.x + threadIdx.x;
    if (idx >= N * EMB) return;
    int i = idx >> 6, k = idx & 63;
    xp[idx] = (k < IN_DIM) ? x[(size_t)i * IN_DIM + k] : 0.0f;
}

// ---------------- weight packing: fp32 W[K,C] -> bf16 B-fragment layout -----
// Packed layout: frag(kt, tileN) is 32 lanes x 16 bf16 (32B per lane,
// contiguous), so the GEMM loads each lane's fragment as one v16bf.
// B fragment element mapping (16-bit B 32x16, ISA 7.12.2):
//   lane = hi*16 + n ; element pair {2v,2v+1} = K rows {kt*32 + 2v + hi*8 + (v>=4 ? 8 : 0), +1}
template<int K, int C>
__global__ void pack_w_bf16(const float* __restrict__ W, __bf16* __restrict__ out) {
    constexpr int KT = (K + 31) / 32;
    constexpr int CT = C / 16;
    int idx = blockIdx.x * blockDim.x + threadIdx.x;    // one thread = one lane-frag
    if (idx >= KT * CT * 32) return;
    int lane  = idx & 31;
    int t     = idx >> 5;
    int tileN = t % CT;
    int kt    = t / CT;
    int n  = lane & 15;
    int hi = lane >> 4;
    __bf16* dst = out + ((size_t)(kt * CT + tileN) * 32 + lane) * 16;
#pragma unroll
    for (int v = 0; v < 8; ++v) {
        int kb = kt * 32 + 2 * v + hi * 8 + (v >= 4 ? 8 : 0);
        float f0 = (kb     < K) ? W[(size_t)kb       * C + tileN * 16 + n] : 0.0f;
        float f1 = (kb + 1 < K) ? W[(size_t)(kb + 1) * C + tileN * 16 + n] : 0.0f;
        dst[2 * v]     = (__bf16)f0;
        dst[2 * v + 1] = (__bf16)f1;
    }
}

// ---------------- WMMA GEMM: out[M,C] = A[M,K] @ W[K,C] + bias ----------------
// One wave owns a 16-row band and ALL C/16 column tiles: A fragment is loaded
// once per k-step (4x b128) and fed to CT independent WMMAs.
// Requires: K % 32 == 0, A rows 16B-aligned (K multiple of 4 floats + aligned base).
template<int K, int C>
__global__ void gemm_bf16_wmma(const float* __restrict__ A,
                               const __bf16* __restrict__ Wp,   // packed fragments
                               const float* __restrict__ bias,
                               float* __restrict__ out, int M) {
    static_assert(K % 32 == 0, "K must be a multiple of 32");
    constexpr int KT = K / 32;
    constexpr int CT = C / 16;
    const int lane  = threadIdx.x;          // 0..31
    const int m     = lane & 15;
    const int hi    = lane >> 4;            // 0 or 1
    const int tileM = blockIdx.x * blockDim.y + threadIdx.y;
    const int row0  = tileM * 16;
    if (row0 >= M) return;                  // wave-uniform exit (EXEC stays all-1)

    v8f acc[CT];
#pragma unroll
    for (int ct = 0; ct < CT; ++ct) {
        v8f z = {};
        acc[ct] = z;
    }
    const float4* arow = (const float4*)(A + (size_t)(row0 + m) * K);

#pragma unroll
    for (int kt = 0; kt < KT; ++kt) {
        // A fragment (16-bit A 16x32, ISA 7.12.2): per lane-half hi the 16
        // elements are two contiguous 8-float runs: [hi*8..+7], [16+hi*8..+7].
        const float4* ap = arow + kt * 8;   // 32 floats = 8 float4
        float4 q0 = ap[hi * 2 + 0];
        float4 q1 = ap[hi * 2 + 1];
        float4 q2 = ap[4 + hi * 2 + 0];
        float4 q3 = ap[4 + hi * 2 + 1];
        v16bf a;
        a[0]  = (__bf16)q0.x; a[1]  = (__bf16)q0.y;
        a[2]  = (__bf16)q0.z; a[3]  = (__bf16)q0.w;
        a[4]  = (__bf16)q1.x; a[5]  = (__bf16)q1.y;
        a[6]  = (__bf16)q1.z; a[7]  = (__bf16)q1.w;
        a[8]  = (__bf16)q2.x; a[9]  = (__bf16)q2.y;
        a[10] = (__bf16)q2.z; a[11] = (__bf16)q2.w;
        a[12] = (__bf16)q3.x; a[13] = (__bf16)q3.y;
        a[14] = (__bf16)q3.z; a[15] = (__bf16)q3.w;

        const __bf16* wl = Wp + ((size_t)kt * CT * 32 + lane) * 16;
#pragma unroll
        for (int ct = 0; ct < CT; ++ct) {
            v16bf b = *(const v16bf*)(wl + (size_t)ct * 32 * 16);
            acc[ct] = __builtin_amdgcn_wmma_f32_16x16x32_bf16(
                /*neg_a=*/false, a, /*neg_b=*/false, b,
                /*c_mod=*/(short)0, acc[ct], /*reuse_a=*/false, /*reuse_b=*/false);
        }
    }

#pragma unroll
    for (int ct = 0; ct < CT; ++ct) {
        float bv = bias[ct * 16 + m];       // column bias (n == lane&15 == m)
#pragma unroll
        for (int r = 0; r < 8; ++r) {       // C layout: VGPR r -> row r (+8 hi)
            out[(size_t)(row0 + r + hi * 8) * C + ct * 16 + m] = acc[ct][r] + bv;
        }
    }
}

// ---------------- h += vn[batch] ----------------
__global__ void add_vn_kernel(float* __restrict__ h, const float* __restrict__ vn,
                              const int* __restrict__ batch, int N) {
    int idx = blockIdx.x * blockDim.x + threadIdx.x;
    if (idx >= N * EMB) return;
    int i = idx >> 6, c = idx & 63;
    h[idx] += vn[batch[i] * EMB + c];
}

// ---------------- fused edge message: agg[dst] += relu(h[src] + ea@We + be) --
// 256 threads = 4 edges x 64 channels. We/be staged in LDS.
__global__ void edge_message_kernel(const float* __restrict__ h,
                                    const float* __restrict__ edge_attr,
                                    const int* __restrict__ src,
                                    const int* __restrict__ dst,
                                    const float* __restrict__ We,   // [10,64]
                                    const float* __restrict__ be,   // [64]
                                    float* __restrict__ agg, int E) {
    __shared__ float sWe[10 * EMB];
    __shared__ float sbe[EMB];
    __shared__ float sEA[4][10];
    const int tid = threadIdx.x;
    for (int i = tid; i < 10 * EMB; i += 256) sWe[i] = We[i];
    if (tid < EMB) sbe[tid] = be[tid];
    const int el = tid >> 6;
    const int c  = tid & 63;
    const int e  = blockIdx.x * 4 + el;
    if (c < 10 && e < E) sEA[el][c] = edge_attr[(size_t)e * 10 + c];
    __syncthreads();
    if (e >= E) return;
    float val = sbe[c];
#pragma unroll
    for (int j = 0; j < 10; ++j) val += sEA[el][j] * sWe[j * EMB + c];
    float msg = h[(size_t)src[e] * EMB + c] + val;
    if (msg > 0.0f)                          // relu: skip zero contributions
        atomicAdd(&agg[(size_t)dst[e] * EMB + c], msg);
}

// ---------------- z = (1+eps[l])*h + agg (in place into agg) ----------------
__global__ void combine_z_kernel(float* __restrict__ agg, const float* __restrict__ h,
                                 const float* __restrict__ eps, int l, int n) {
    int idx = blockIdx.x * blockDim.x + threadIdx.x;
    if (idx >= n) return;
    agg[idx] = (1.0f + eps[l]) * h[idx] + agg[idx];
}

// ---------------- BN column stats: sums[c]=Σx, sums[C+c]=Σx² ----------------
__global__ void bn_stats_kernel(const float* __restrict__ X, float* __restrict__ sums,
                                int Nrows, int C) {
    const int c = threadIdx.x;               // blockDim.x == C
    float s = 0.0f, ss = 0.0f;
    for (int r = blockIdx.x * blockDim.y + threadIdx.y; r < Nrows;
         r += gridDim.x * blockDim.y) {
        float v = X[(size_t)r * C + c];
        s += v; ss += v * v;
    }
    atomicAdd(&sums[c], s);
    atomicAdd(&sums[C + c], ss);
}

// ---------------- BN apply: y = (x-mu)*rsqrt(var+eps) + 1e-4 (+relu) -------
__global__ void bn_apply_kernel(const float* __restrict__ X, float* __restrict__ Y,
                                const float* __restrict__ sums, int Nrows, int C,
                                int doRelu) {
    int idx = blockIdx.x * blockDim.x + threadIdx.x;
    if (idx >= Nrows * C) return;
    int c = idx % C;
    float inv = 1.0f / (float)Nrows;
    float mu  = sums[c] * inv;
    float var = sums[C + c] * inv - mu * mu;
    float y   = (X[idx] - mu) * rsqrtf(var + 1e-5f) + 1e-4f;
    if (doRelu) y = fmaxf(y, 0.0f);
    Y[idx] = y;
}

// ---------------- segment_sum(h, batch) accumulated into vtsum --------------
__global__ void seg_sum_kernel(const float* __restrict__ h, const int* __restrict__ batch,
                               float* __restrict__ vtsum, int N) {
    int idx = blockIdx.x * blockDim.x + threadIdx.x;
    if (idx >= N * EMB) return;
    int i = idx >> 6, c = idx & 63;
    atomicAdd(&vtsum[batch[i] * EMB + c], h[idx]);
}

// ---------------- node attention head ----------------
__global__ void node_att_kernel(const float* __restrict__ nx,
                                const float* __restrict__ Wna,   // [64,2]
                                const float* __restrict__ bna,
                                float* __restrict__ out, int N) {
    __shared__ float sW[EMB * 2];
    __shared__ float sb[2];
    int tid = threadIdx.x;
    if (tid < EMB * 2) sW[tid] = Wna[tid];
    if (tid < 2) sb[tid] = bna[tid];
    __syncthreads();
    int i = blockIdx.x * blockDim.x + tid;
    if (i >= N) return;
    const float* r = nx + (size_t)i * EMB;
    float z0 = sb[0], z1 = sb[1];
#pragma unroll 8
    for (int k = 0; k < EMB; ++k) {
        float v = r[k];
        z0 += v * sW[k * 2];
        z1 += v * sW[k * 2 + 1];
    }
    float mx = fmaxf(z0, z1);
    float e0 = __expf(z0 - mx), e1 = __expf(z1 - mx);
    out[i] = e1 / (e0 + e1);
}

// ---------------- edge attention head ----------------
__global__ void edge_att_kernel(const float* __restrict__ nx,
                                const int* __restrict__ src,
                                const int* __restrict__ dst,
                                const float* __restrict__ Wea,   // [128,2]
                                const float* __restrict__ bea,
                                float* __restrict__ out, int E) {
    __shared__ float sW[2 * EMB * 2];
    __shared__ float sb[2];
    int tid = threadIdx.x;
    if (tid < 2 * EMB * 2) sW[tid] = Wea[tid];
    if (tid < 2) sb[tid] = bea[tid];
    __syncthreads();
    int e = blockIdx.x * blockDim.x + tid;
    if (e >= E) return;
    const float* rs = nx + (size_t)src[e] * EMB;
    const float* rd = nx + (size_t)dst[e] * EMB;
    float z0 = sb[0], z1 = sb[1];
#pragma unroll 8
    for (int k = 0; k < EMB; ++k) {
        float v = rs[k];
        z0 += v * sW[k * 2];
        z1 += v * sW[k * 2 + 1];
    }
#pragma unroll 8
    for (int k = 0; k < EMB; ++k) {
        float v = rd[k];
        z0 += v * sW[(EMB + k) * 2];
        z1 += v * sW[(EMB + k) * 2 + 1];
    }
    float mx = fmaxf(z0, z1);
    float e0 = __expf(z0 - mx), e1 = __expf(z1 - mx);
    out[e] = e1 / (e0 + e1);
}

// ---------------------------------------------------------------------------
extern "C" void kernel_launch(void* const* d_in, const int* in_sizes, int n_in,
                              void* d_out, int out_size, void* d_ws, size_t ws_size,
                              hipStream_t stream) {
    (void)n_in; (void)out_size; (void)ws_size;
    const float* x         = (const float*)d_in[0];
    const float* edge_attr = (const float*)d_in[1];
    const int*   ei        = (const int*)  d_in[2];
    const int*   batch     = (const int*)  d_in[3];
    const float* Wenc      = (const float*)d_in[4];
    const float* benc      = (const float*)d_in[5];
    const float* We        = (const float*)d_in[6];
    const float* be        = (const float*)d_in[7];
    const float* eps       = (const float*)d_in[8];
    const float* W1        = (const float*)d_in[9];
    const float* b1        = (const float*)d_in[10];
    const float* W2        = (const float*)d_in[11];
    const float* b2        = (const float*)d_in[12];
    const float* Wv1       = (const float*)d_in[13];
    const float* bv1       = (const float*)d_in[14];
    const float* Wv2       = (const float*)d_in[15];
    const float* bv2       = (const float*)d_in[16];
    const float* Wea       = (const float*)d_in[17];
    const float* bea       = (const float*)d_in[18];
    const float* Wna       = (const float*)d_in[19];
    const float* bna       = (const float*)d_in[20];

    const int N = in_sizes[0] / IN_DIM;  // 50000
    const int E = in_sizes[1] / 10;      // 800000
    const int* src = ei;
    const int* dst = ei + E;

    // ---- workspace layout (floats; d_ws is 256B aligned) ----
    float* ws    = (float*)d_ws;
    float* h     = ws;                 size_t o = (size_t)N * EMB;
    float* agg   = ws + o;             o += (size_t)N * EMB;
    float* t1    = ws + o;             o += (size_t)N * 2 * EMB;
    float* z2    = ws + o;             o += (size_t)N * EMB;
    float* xp    = ws + o;             o += (size_t)N * EMB;   // padded x
    float* stats = ws + o;             o += 256;
    float* vstat = ws + o;             o += 256;
    float* vn    = ws + o;             o += GRAPHS * EMB;
    float* vtsum = ws + o;             o += GRAPHS * EMB;
    float* vtt   = ws + o;             o += GRAPHS * 2 * EMB;
    float* vtmp  = ws + o;             o += GRAPHS * EMB;
    o = (o + 7) & ~(size_t)7;          // 32B align for v16bf loads
    __bf16* pwA  = (__bf16*)(ws + o);  o += 4096;   // up to 8192 bf16
    __bf16* pwB  = (__bf16*)(ws + o);  o += 4096;
    float* outp  = (float*)d_out;

    auto zero = [&](float* p, int n) {
        fill_zero<<<(n + 255) / 256, 256, 0, stream>>>(p, n);
    };
    auto bn = [&](const float* X, float* Y, float* st, int Nrows, int C, int relu) {
        zero(st, 2 * C);
        dim3 block(C, 256 / C);
        int rowBlocks = (Nrows + (int)block.y - 1) / (int)block.y;
        dim3 grid(rowBlocks < 256 ? rowBlocks : 256);
        bn_stats_kernel<<<grid, block, 0, stream>>>(X, st, Nrows, C);
        bn_apply_kernel<<<((Nrows * C) + 255) / 256, 256, 0, stream>>>(X, Y, st, Nrows, C, relu);
    };
    auto gemmGrid = [&](int M) {
        int tilesM = (M + 15) / 16;
        return dim3((tilesM + 3) / 4);
    };
    const dim3 gblock(32, 4);

    // vn = 0
    zero(vn, GRAPHS * EMB);

    // h = x @ Wenc + benc   (x zero-padded to K=64; packed W rows >=39 are 0)
    pad_x_kernel<<<(N * EMB + 255) / 256, 256, 0, stream>>>(x, xp, N);
    pack_w_bf16<IN_DIM, 64><<<1, 256, 0, stream>>>(Wenc, pwA);
    gemm_bf16_wmma<64, 64><<<gemmGrid(N), gblock, 0, stream>>>(xp, pwA, benc, h, N);

    for (int l = 0; l < 3; ++l) {
        // h += vn[batch]  (h becomes h_in)
        add_vn_kernel<<<(N * EMB + 255) / 256, 256, 0, stream>>>(h, vn, batch, N);
        // agg = segment_sum(relu(h[src] + edge_attr@We[l] + be[l]), dst)
        zero(agg, N * EMB);
        edge_message_kernel<<<(E + 3) / 4, 256, 0, stream>>>(
            h, edge_attr, src, dst, We + (size_t)l * 10 * EMB, be + (size_t)l * EMB, agg, E);
        // z = (1+eps[l])*h + agg  (in place in agg)
        combine_z_kernel<<<(N * EMB + 255) / 256, 256, 0, stream>>>(agg, h, eps, l, N * EMB);

        // t1 = relu(bn(z @ W1 + b1))
        pack_w_bf16<64, 128><<<2, 256, 0, stream>>>(W1 + (size_t)l * EMB * 2 * EMB, pwA);
        gemm_bf16_wmma<64, 128><<<gemmGrid(N), gblock, 0, stream>>>(
            agg, pwA, b1 + (size_t)l * 2 * EMB, t1, N);
        bn(t1, t1, stats, N, 2 * EMB, 1);

        // z2 = t1 @ W2 + b2
        pack_w_bf16<128, 64><<<2, 256, 0, stream>>>(W2 + (size_t)l * 2 * EMB * EMB, pwB);
        gemm_bf16_wmma<128, 64><<<gemmGrid(N), gblock, 0, stream>>>(
            t1, pwB, b2 + (size_t)l * EMB, z2, N);

        if (l < 2) {
            // virtual-node update (uses h_in, before h is overwritten)
            copy_buf<<<(GRAPHS * EMB + 255) / 256, 256, 0, stream>>>(vtsum, vn, GRAPHS * EMB);
            seg_sum_kernel<<<(N * EMB + 255) / 256, 256, 0, stream>>>(h, batch, vtsum, N);
            pack_w_bf16<64, 128><<<2, 256, 0, stream>>>(Wv1 + (size_t)l * EMB * 2 * EMB, pwA);
            gemm_bf16_wmma<64, 128><<<gemmGrid(GRAPHS), gblock, 0, stream>>>(
                vtsum, pwA, bv1 + (size_t)l * 2 * EMB, vtt, GRAPHS);
            bn(vtt, vtt, vstat, GRAPHS, 2 * EMB, 1);
            pack_w_bf16<128, 64><<<2, 256, 0, stream>>>(Wv2 + (size_t)l * 2 * EMB * EMB, pwB);
            gemm_bf16_wmma<128, 64><<<gemmGrid(GRAPHS), gblock, 0, stream>>>(
                vtt, pwB, bv2 + (size_t)l * EMB, vtmp, GRAPHS);
            bn(vtmp, vn, vstat, GRAPHS, EMB, 1);
        }
        // h = bn(z2) (+relu if not last layer)
        bn(z2, h, stats, N, EMB, (l < 2) ? 1 : 0);
    }

    // heads
    node_att_kernel<<<(N + 255) / 256, 256, 0, stream>>>(h, Wna, bna, outp, N);
    edge_att_kernel<<<(E + 255) / 256, 256, 0, stream>>>(h, src, dst, Wea, bea, outp + N, E);
}
